// CrossAttentionGenerator_26328149525078
// MI455X (gfx1250) — compile-verified, matrix operand order checked by reading, not run on hardware
//
#include <hip/hip_runtime.h>

typedef __attribute__((ext_vector_type(16))) _Float16 v16h;
typedef __attribute__((ext_vector_type(8)))  _Float16 v8h;
typedef __attribute__((ext_vector_type(4)))  _Float16 v4h;
typedef __attribute__((ext_vector_type(8)))  float    v8f;

namespace {
constexpr int kNTok = 64;
constexpr int kC    = 256;
constexpr int kTP   = 768;   // 3*P
constexpr int kH    = 8;
constexpr int kD    = 32;

// ---- LDS layout, units = _Float16 elements ----
constexpr int QK_LD   = 40;                 // q/k rows [m][d<=31], padded (bank-friendly)
constexpr int VT_LD   = 72;                 // v^T rows [d][m<=63], padded
constexpr int QK_HEAD = kNTok * QK_LD;      // 2560
constexpr int VT_HEAD = kD * VT_LD;         // 2304
constexpr int Q_OFF   = 0;
constexpr int K_OFF   = kH * QK_HEAD;       // 20480
constexpr int V_OFF   = 2 * kH * QK_HEAD;   // 40960
constexpr int SIDE    = V_OFF + kH * VT_HEAD; // 59392 per side (origin / da)
constexpr int OFF_X   = 2 * SIDE;           // 118784
constexpr int XH_LD   = 264;                // normalized tokens [s][n][c], padded
constexpr int PS_LD   = 72;                 // softmax probs [n][m], padded
constexpr int PS_WAVE = kNTok * PS_LD;      // 4608 halves per wave
constexpr int OS_LD   = 72;                 // output staging [dir][ch][n], padded
constexpr int SMEM_HALVES = OFF_X + 8 * PS_WAVE;           // 155648
constexpr size_t SMEM_BYTES = SMEM_HALVES * sizeof(_Float16); // 311296 (<= 320KB WGP LDS)
} // namespace

// Gather one 16-bit WMMA A/B operand from a row-major [row][K] array.
// Caller passes (base + row(lane)*ld + kstep*32). 16-bit operand layout:
// lanes 0-15 hold K = {0..7, 16..23}, lanes 16-31 hold K = {8..15, 24..31}.
__device__ __forceinline__ v16h load_op16(const _Float16* row) {
  const int ko = (threadIdx.x & 16) ? 8 : 0;
  v8h c0 = *(const v8h*)(row + ko);
  v8h c1 = *(const v8h*)(row + ko + 16);
  v16h r;
#pragma unroll
  for (int i = 0; i < 8; ++i) { r[i] = c0[i]; r[i + 8] = c1[i]; }
  return r;
}

__device__ __forceinline__ v8f wmma_f16(v16h a, v16h b, v8f c) {
  return __builtin_amdgcn_wmma_f32_16x16x32_f16(false, a, false, b, (short)0, c,
                                                false, false);
}

// One-time: swizzle W_qkv (fp32 -> f16) into per-lane-contiguous B-operand
// layout [side][ntile(48)][kstep(8)][lane(32)][16], and transpose pos embeds
// to [side][c][n] so the main kernel's global reads are fully coalesced.
__global__ __launch_bounds__(256) void xattn_prep(
    const float* __restrict__ Wo, const float* __restrict__ Wd,
    const float* __restrict__ po, const float* __restrict__ pd,
    _Float16* __restrict__ wB, float* __restrict__ pos_t) {
  const int WSZ = kTP * kC; // 196608 per side
  int idx = blockIdx.x * 256 + threadIdx.x;
  if (idx < 2 * WSZ) {
    int s  = idx / WSZ;
    int r  = idx % WSZ;
    int nt = r >> 12;
    int r2 = r & 4095;
    int kk = r2 >> 9;
    int r3 = r2 & 511;
    int ln = r3 >> 4;
    int h  = r3 & 15;
    int n  = nt * 16 + (ln & 15);
    int kb = (ln & 16) ? 8 : 0;
    int k  = kk * 32 + kb + (h < 8 ? h : (h - 8) + 16);
    const float* W = s ? Wd : Wo;
    wB[idx] = (_Float16)W[n * kC + k];
  } else {
    int j = idx - 2 * WSZ;
    if (j < 2 * kC * kNTok) {
      int s  = j >> 14;
      int cn = j & 16383;
      int c  = cn >> 6;
      int n  = cn & 63;
      const float* p = s ? pd : po;
      pos_t[j] = p[n * kC + c];
    }
  }
}

__global__ __launch_bounds__(256) void xattn_main(
    const float* __restrict__ xo_g, const float* __restrict__ xd_g,
    const float* __restrict__ proto_g,
    const float* __restrict__ nwo, const float* __restrict__ nbo,
    const float* __restrict__ nwd, const float* __restrict__ nbd,
    const float* __restrict__ mixw,
    const _Float16* __restrict__ wB, const float* __restrict__ pos_t,
    float* __restrict__ out) {
  extern __shared__ _Float16 smem[];
  const int tid  = threadIdx.x;
  const int lane = tid & 31;
  const int wv   = tid >> 5;
  const int lrow = lane & 15;                 // row-in-tile for operand loads
  const int lhi  = (lane & 16) ? 8 : 0;       // C-layout row offset for hi half
  const long base = (long)blockIdx.x * (kC * kNTok);
  const v8f vzero = {};

  // ---------- Phase 1a: coalesced load x + pos -> fp32 staging (pad 257) ----
  float* tmp0 = (float*)smem;            // origin tokens [n][c], row stride 257
  float* tmp1 = tmp0 + kNTok * 257;      // da tokens
  for (int i4 = tid; i4 < (kC * kNTok) / 4; i4 += 256) {
    int i = i4 * 4;
    int c = i >> 6, n0 = i & 63;         // 4 consecutive n, same c
    float4 a  = *(const float4*)(xo_g + base + i);
    float4 b  = *(const float4*)(xd_g + base + i);
    float4 pa = *(const float4*)(pos_t + i);
    float4 pb = *(const float4*)(pos_t + kC * kNTok + i);
    tmp0[(n0 + 0) * 257 + c] = a.x + pa.x;
    tmp0[(n0 + 1) * 257 + c] = a.y + pa.y;
    tmp0[(n0 + 2) * 257 + c] = a.z + pa.z;
    tmp0[(n0 + 3) * 257 + c] = a.w + pa.w;
    tmp1[(n0 + 0) * 257 + c] = b.x + pb.x;
    tmp1[(n0 + 1) * 257 + c] = b.y + pb.y;
    tmp1[(n0 + 2) * 257 + c] = b.z + pb.z;
    tmp1[(n0 + 3) * 257 + c] = b.w + pb.w;
  }
  __syncthreads();

  // ---------- Phase 1b: layernorm over C=256 (wave32 shfl reductions) ------
  _Float16* xh = smem + OFF_X;           // [s][n][XH_LD] f16
  for (int s = 0; s < 2; ++s) {
    const float* tmp = s ? tmp1 : tmp0;
    const float* nw  = s ? nwd : nwo;
    const float* nb  = s ? nbd : nbo;
    float wreg[8], breg[8];
#pragma unroll
    for (int j = 0; j < 8; ++j) { wreg[j] = nw[lane + 32 * j]; breg[j] = nb[lane + 32 * j]; }
    for (int i = 0; i < 8; ++i) {
      int n = wv * 8 + i;
      float v[8]; float sum = 0.f;
#pragma unroll
      for (int j = 0; j < 8; ++j) { v[j] = tmp[n * 257 + lane + 32 * j]; sum += v[j]; }
#pragma unroll
      for (int m = 1; m < 32; m <<= 1) sum += __shfl_xor(sum, m, 32);
      float u = sum * (1.f / 256.f);
      float var = 0.f;
#pragma unroll
      for (int j = 0; j < 8; ++j) { float d = v[j] - u; var += d * d; }
#pragma unroll
      for (int m = 1; m < 32; m <<= 1) var += __shfl_xor(var, m, 32);
      float rs = rsqrtf(var * (1.f / 256.f) + 1e-5f);
#pragma unroll
      for (int j = 0; j < 8; ++j) {
        int c = lane + 32 * j;
        xh[(s * kNTok + n) * XH_LD + c] = (_Float16)(wreg[j] * ((v[j] - u) * rs) + breg[j]);
      }
    }
  }
  __syncthreads();  // qkv region (was fp32 staging) is now free

  // ---------- Phase 2: QKV = X(64x256) * W^T(256x768), both sides ----------
  {
    const int s  = wv >> 2;              // side
    const int mt = wv & 3;               // M tile
    const int nbase = mt * 16 + lhi;
    const _Float16* xrow = xh + (s * kNTok + mt * 16 + lrow) * XH_LD;
    v16h a[8];
#pragma unroll
    for (int kk = 0; kk < 8; ++kk) a[kk] = load_op16(xrow + kk * 32);

    // nt -> destination is wave-uniform: t = nt>>4 (q/k/v), h = (nt>>1)&7,
    // dd = (nt&1)*16 + lane_row.  Scalar branch, no per-lane predication.
    auto store_tile = [&](int nt, const v8f& acc) {
      const int t  = nt >> 4;
      const int h  = (nt >> 1) & 7;
      const int dd = ((nt & 1) << 4) + lrow;
      if (t < 2) {
        _Float16* dst = smem + s * SIDE + (t == 0 ? Q_OFF : K_OFF) + h * QK_HEAD + dd;
#pragma unroll
        for (int r = 0; r < 8; ++r) dst[(nbase + r) * QK_LD] = (_Float16)acc[r];
      } else {
        // v^T[d][m]: 8 consecutive m per lane -> one b128 store
        _Float16* dst = smem + s * SIDE + V_OFF + h * VT_HEAD + dd * VT_LD + nbase;
        v8h hv;
#pragma unroll
        for (int r = 0; r < 8; ++r) hv[r] = (_Float16)acc[r];
        *(v8h*)dst = hv;
      }
    };

    for (int nt = 0; nt < 48; nt += 2) {   // 2 tiles -> 2 independent WMMA chains
      v8f acc0 = vzero, acc1 = vzero;
      const _Float16* bp0 = wB + (size_t)(s * 48 + nt) * 4096 + lane * 16;
#pragma unroll
      for (int kk = 0; kk < 8; ++kk) {
        v16h b0 = *(const v16h*)(bp0 + kk * 512);
        v16h b1 = *(const v16h*)(bp0 + 4096 + kk * 512);
        acc0 = wmma_f16(a[kk], b0, acc0);
        acc1 = wmma_f16(a[kk], b1, acc1);
      }
      store_tile(nt, acc0);
      store_tile(nt + 1, acc1);
    }
  }
  __syncthreads();

  // ---------- Phase 3: cross attention, wave = head, both directions -------
  v8f o[2][8];                           // O accumulators (64x32 per dir)
  {
    const float scale = 0.17677669529663688f;  // 1/sqrt(32)
    _Float16* ps = smem + OFF_X + wv * PS_WAVE; // per-wave prob tile [n][m]
#pragma unroll
    for (int dir = 0; dir < 2; ++dir) {
      // dir 0: q_origin x k_da -> * v_da  (= da_v,     mix_w[1])
      // dir 1: q_da x k_origin -> * v_o   (= origin_v, mix_w[0])
      const int sq = dir, skv = 1 - dir;
      const _Float16* qp = smem + sq  * SIDE + Q_OFF + wv * QK_HEAD;
      const _Float16* kp = smem + skv * SIDE + K_OFF + wv * QK_HEAD;
      const _Float16* vp = smem + skv * SIDE + V_OFF + wv * VT_HEAD;
      v16h bk[4];
#pragma unroll
      for (int ct = 0; ct < 4; ++ct)
        bk[ct] = load_op16(kp + (ct * 16 + lrow) * QK_LD);
#pragma unroll
      for (int mt = 0; mt < 4; ++mt) {
        v16h aq = load_op16(qp + (mt * 16 + lrow) * QK_LD);
        v8f S[4] = {vzero, vzero, vzero, vzero};
#pragma unroll
        for (int ct = 0; ct < 4; ++ct) S[ct] = wmma_f16(aq, bk[ct], S[ct]);
        // row softmax: rows of a C tile live in one 16-lane half (xor 1,2,4,8)
#pragma unroll
        for (int r = 0; r < 8; ++r) {
          float mx = fmaxf(fmaxf(S[0][r], S[1][r]), fmaxf(S[2][r], S[3][r]));
#pragma unroll
          for (int m = 1; m <= 8; m <<= 1) mx = fmaxf(mx, __shfl_xor(mx, m, 32));
          float p0 = __expf((S[0][r] - mx) * scale);
          float p1 = __expf((S[1][r] - mx) * scale);
          float p2 = __expf((S[2][r] - mx) * scale);
          float p3 = __expf((S[3][r] - mx) * scale);
          float sum = p0 + p1 + p2 + p3;
#pragma unroll
          for (int m = 1; m <= 8; m <<= 1) sum += __shfl_xor(sum, m, 32);
          float inv = 1.f / sum;
          int n = mt * 16 + r + lhi;
          _Float16* pr = ps + n * PS_LD + lrow;
          pr[0]  = (_Float16)(p0 * inv);
          pr[16] = (_Float16)(p1 * inv);
          pr[32] = (_Float16)(p2 * inv);
          pr[48] = (_Float16)(p3 * inv);
        }
      }
      // O = P(64x64) * V(64x32), K split into 2 steps of 32
      v16h bv[2][2];
#pragma unroll
      for (int kk = 0; kk < 2; ++kk)
#pragma unroll
        for (int nt2 = 0; nt2 < 2; ++nt2)
          bv[kk][nt2] = load_op16(vp + (nt2 * 16 + lrow) * VT_LD + kk * 32);
#pragma unroll
      for (int mt = 0; mt < 4; ++mt) {
        v16h ap0 = load_op16(ps + (mt * 16 + lrow) * PS_LD);
        v16h ap1 = load_op16(ps + (mt * 16 + lrow) * PS_LD + 32);
#pragma unroll
        for (int nt2 = 0; nt2 < 2; ++nt2) {
          v8f acc = vzero;
          acc = wmma_f16(ap0, bv[0][nt2], acc);
          acc = wmma_f16(ap1, bv[1][nt2], acc);
          o[dir][mt * 2 + nt2] = acc;
        }
      }
    }
  }
  __syncthreads();  // all q/k/v consumed -> reuse region for output staging

  // ---------- stage O into LDS: [dir][ch][n] (pad 72), b128 stores ---------
  {
    _Float16* os = smem;
#pragma unroll
    for (int dir = 0; dir < 2; ++dir)
#pragma unroll
      for (int mt = 0; mt < 4; ++mt)
#pragma unroll
        for (int nt2 = 0; nt2 < 2; ++nt2) {
          int ch = wv * 32 + nt2 * 16 + lrow;
          int nbase = mt * 16 + lhi;
          v8h hv;
#pragma unroll
          for (int r = 0; r < 8; ++r) hv[r] = (_Float16)o[dir][mt * 2 + nt2][r];
          *(v8h*)(os + (dir * kC + ch) * OS_LD + nbase) = hv;
        }
  }
  __syncthreads();

  // ---------- Phase 4: mix with prototypes, coalesced store ----------------
  {
    const float m0 = mixw[0], m1 = mixw[1], m2 = mixw[2];
    const _Float16* os = smem;
    for (int i4 = tid; i4 < (kC * kNTok) / 4; i4 += 256) {
      int i = i4 * 4;
      int c = i >> 6, n0 = i & 63;
      float4 pr = *(const float4*)(proto_g + base + i);
      v4h ov = *(const v4h*)(os + (kC + c) * OS_LD + n0);  // origin_v (dir 1)
      v4h dv = *(const v4h*)(os + c * OS_LD + n0);         // da_v     (dir 0)
      float4 r;
      r.x = m0 * (float)ov[0] + m1 * (float)dv[0] + m2 * pr.x;
      r.y = m0 * (float)ov[1] + m1 * (float)dv[1] + m2 * pr.y;
      r.z = m0 * (float)ov[2] + m1 * (float)dv[2] + m2 * pr.z;
      r.w = m0 * (float)ov[3] + m1 * (float)dv[3] + m2 * pr.w;
      *(float4*)(out + base + i) = r;
    }
  }
}

extern "C" void kernel_launch(void* const* d_in, const int* in_sizes, int n_in,
                              void* d_out, int out_size, void* d_ws, size_t ws_size,
                              hipStream_t stream) {
  const float* xo   = (const float*)d_in[0];
  const float* xd   = (const float*)d_in[1];
  const float* xp   = (const float*)d_in[2];
  const float* po   = (const float*)d_in[3];
  const float* pd   = (const float*)d_in[4];
  const float* nwo  = (const float*)d_in[5];
  const float* nbo  = (const float*)d_in[6];
  const float* nwd  = (const float*)d_in[7];
  const float* nbd  = (const float*)d_in[8];
  const float* Wo   = (const float*)d_in[9];
  const float* Wd   = (const float*)d_in[10];
  const float* mixw = (const float*)d_in[11];
  float* out = (float*)d_out;

  // workspace: [0, 768KB) swizzled f16 weights; then 128KB transposed pos
  _Float16* wB = (_Float16*)d_ws;
  float* pos_t = (float*)((char*)d_ws + (size_t)2 * kTP * kC * sizeof(_Float16));

  int prep_elems = 2 * kTP * kC + 2 * kC * kNTok; // 425984
  xattn_prep<<<(prep_elems + 255) / 256, 256, 0, stream>>>(Wo, Wd, po, pd, wB, pos_t);

  (void)hipFuncSetAttribute((const void*)xattn_main,
                            hipFuncAttributeMaxDynamicSharedMemorySize,
                            (int)SMEM_BYTES);
  int nbatch = in_sizes[0] / (kC * kNTok); // 2048
  xattn_main<<<nbatch, 256, SMEM_BYTES, stream>>>(xo, xd, xp, nwo, nbo, nwd, nbd,
                                                  mixw, wB, pos_t, out);
  (void)n_in; (void)out_size; (void)ws_size;
}
